// NonLocalGaussianAttention_38783554683410
// MI455X (gfx1250) — matrix operand order repaired and checked by reading, not compiled, not run on hardware
//
#include <hip/hip_runtime.h>

typedef __attribute__((ext_vector_type(16))) __bf16 v16bf;
typedef __attribute__((ext_vector_type(8)))  float  v8f;
typedef unsigned int u32x4 __attribute__((ext_vector_type(4)));
typedef unsigned int u32x8 __attribute__((ext_vector_type(8)));

#define BB 8
#define CC 64
#define C2 32
#define NN 4096
#define NCHUNK 128            // NN / 32 key-chunks per batch
#define CHUNK_SHORTS 3072     // [K 32*32 | Vt 64*32] bf16 = 6 KB
#define CHUNK_BYTES  6144
#define CHUNK_DWORDS 1536

__device__ inline unsigned short f2bf(float f) {
    unsigned int u = __builtin_bit_cast(unsigned int, f);
    unsigned int r = (u + 0x7FFFu + ((u >> 16) & 1u)) >> 16;
    return (unsigned short)r;
}

// Low 32 bits of a generic pointer to __shared__ = LDS byte offset.
__device__ inline unsigned lds_off(const void* p) {
    return (unsigned)(unsigned long long)p;
}

// Tensor Data Mover: 1-D async copy of CHUNK_BYTES from global to LDS.
// D# group0/group1 per CDNA5 ISA; 2-operand
// form (VADDR2/3 NULL -> tensors up to 2D). Tracked by TENSORcnt.
__device__ inline void tdm_load_chunk(unsigned lds_addr, unsigned long long gaddr,
                                      u32x8 g1) {
    u32x4 g0;
    g0.x = 1u;                                            // count=1, user mode
    g0.y = lds_addr;                                      // lds_addr[63:32]
    g0.z = (unsigned)(gaddr & 0xFFFFFFFFull);             // global_addr lo
    g0.w = (unsigned)((gaddr >> 32) & 0x01FFFFFFull)      // global_addr hi
         | (2u << 30);                                    // type=2 ("image")
    asm volatile("tensor_load_to_lds %0, %1" :: "s"(g0), "s"(g1) : "memory");
}

__device__ inline u32x8 tdm_group1(void) {
    u32x8 g1;
    g1[0] = (2u << 16);                        // data_size = 4B dwords
    g1[1] = (CHUNK_DWORDS & 0xFFFFu) << 16;    // tensor_dim0[15:0]
    g1[2] = (CHUNK_DWORDS >> 16) | (1u << 16); // tensor_dim0 hi | tensor_dim1=1 lo
    g1[3] = ((unsigned)CHUNK_DWORDS) << 16;    // tensor_dim1 hi=0 | tile_dim0=1536
    g1[4] = 1u;                                // tile_dim1=1, tile_dim2=0
    g1[5] = CHUNK_DWORDS;                      // tensor_dim0_stride lo
    g1[6] = 0u;                                // stride hi | tensor_dim1_stride lo
    g1[7] = 0u;
    return g1;
}

// ---------------------------------------------------------------------------
// Projection: q/k/v = 1x1 convs. Writes Q[B][N][32] bf16, chunked KV blocks
// ([K 32x32][Vt 64x32] bf16 per 32-key chunk, TDM friendly), ksq[B][N] f32.
// ---------------------------------------------------------------------------
__global__ __launch_bounds__(256) void proj_kernel(
    const float* __restrict__ x,
    const float* __restrict__ Wq, const float* __restrict__ bq,
    const float* __restrict__ Wk, const float* __restrict__ bk,
    const float* __restrict__ Wv, const float* __restrict__ bv,
    unsigned short* __restrict__ Qo, unsigned short* __restrict__ KV,
    float* __restrict__ Ksq)
{
    __shared__ float sWqT[CC * C2];   // [c][o]
    __shared__ float sWkT[CC * C2];   // [c][o]
    __shared__ float sWvT[CC * CC];   // [c][o]
    __shared__ float sB[C2 + C2 + CC];

    const int tid = threadIdx.x;

    for (int i = tid; i < CC * C2; i += 256) {
        int o = i / CC, c = i % CC;
        sWqT[c * C2 + o] = Wq[i];
        sWkT[c * C2 + o] = Wk[i];
    }
    for (int i = tid; i < CC * CC; i += 256) {
        int o = i / CC, c = i % CC;
        sWvT[c * CC + o] = Wv[i];
    }
    if (tid < C2)               sB[tid] = bq[tid];
    else if (tid < 2 * C2)      sB[tid] = bk[tid - C2];
    else if (tid < 2 * C2 + CC) sB[tid] = bv[tid - 2 * C2];
    __syncthreads();

    const int b = blockIdx.x >> 4;                    // 16 blocks per batch
    const int n = ((blockIdx.x & 15) << 8) + tid;     // pixel index

    float q[C2], k[C2], v[CC];
#pragma unroll
    for (int o = 0; o < C2; ++o) { q[o] = sB[o]; k[o] = sB[C2 + o]; }
#pragma unroll
    for (int o = 0; o < CC; ++o) { v[o] = sB[2 * C2 + o]; }

    const float* xb = x + (size_t)b * CC * NN + n;
    for (int c = 0; c < CC; ++c) {
        float xc = xb[(size_t)c * NN];
        const float4* wq4 = (const float4*)(sWqT + c * C2);
        const float4* wk4 = (const float4*)(sWkT + c * C2);
        const float4* wv4 = (const float4*)(sWvT + c * CC);
#pragma unroll
        for (int o4 = 0; o4 < C2 / 4; ++o4) {
            float4 w = wq4[o4];
            q[4 * o4 + 0] += w.x * xc; q[4 * o4 + 1] += w.y * xc;
            q[4 * o4 + 2] += w.z * xc; q[4 * o4 + 3] += w.w * xc;
        }
#pragma unroll
        for (int o4 = 0; o4 < C2 / 4; ++o4) {
            float4 w = wk4[o4];
            k[4 * o4 + 0] += w.x * xc; k[4 * o4 + 1] += w.y * xc;
            k[4 * o4 + 2] += w.z * xc; k[4 * o4 + 3] += w.w * xc;
        }
#pragma unroll
        for (int o4 = 0; o4 < CC / 4; ++o4) {
            float4 w = wv4[o4];
            v[4 * o4 + 0] += w.x * xc; v[4 * o4 + 1] += w.y * xc;
            v[4 * o4 + 2] += w.z * xc; v[4 * o4 + 3] += w.w * xc;
        }
    }

    float ks = 0.0f;
#pragma unroll
    for (int o = 0; o < C2; ++o) ks += k[o] * k[o];

    const int chunk = n >> 5;       // 32-key chunk within batch
    const int kic   = n & 31;       // key index in chunk
    unsigned short* kvb = KV + ((size_t)b * NCHUNK + chunk) * CHUNK_SHORTS;

    unsigned int* Qp = (unsigned int*)(Qo + ((size_t)b * NN + n) * C2);
#pragma unroll
    for (int o = 0; o < C2 / 2; ++o)
        Qp[o] = (unsigned int)f2bf(q[2 * o]) | ((unsigned int)f2bf(q[2 * o + 1]) << 16);

    unsigned int* Kp = (unsigned int*)(kvb + kic * C2);      // K: [key][32ch]
#pragma unroll
    for (int o = 0; o < C2 / 2; ++o)
        Kp[o] = (unsigned int)f2bf(k[2 * o]) | ((unsigned int)f2bf(k[2 * o + 1]) << 16);

#pragma unroll
    for (int o = 0; o < CC; ++o)                             // Vt: [ch][key32]
        kvb[1024 + o * 32 + kic] = f2bf(v[o]);

    Ksq[(size_t)b * NN + n] = ks;
}

// ---------------------------------------------------------------------------
// Fused flash-attention: logit = (qk - 0.5*ksq)/sigma^2 (q_sq cancels in
// softmax). One wave = 16 query rows, block = 128. K/V chunks streamed by
// the Tensor Data Mover with double-buffered LDS.
// ---------------------------------------------------------------------------
__global__ __launch_bounds__(256) void attn_kernel(
    const float* __restrict__ x,
    const unsigned short* __restrict__ Q,
    const unsigned short* __restrict__ KV,
    const float* __restrict__ Ksq,
    const float* __restrict__ sigma_p,
    float* __restrict__ out)
{
    __shared__ unsigned short sKV[2][CHUNK_SHORTS];  // double-buffered 6KB chunks
    __shared__ unsigned short sP[8][16 * 32];        // per-wave P staging

    const int tid  = threadIdx.x;
    const int lane = tid & 31;
    const int wave = tid >> 5;
    const int h    = lane >> 4;     // half: selects K-range per WMMA layouts
    const int cl   = lane & 15;     // column / row-within-half

    const int b  = blockIdx.x >> 5;                        // 32 blocks/batch
    const int q0 = (blockIdx.x & 31) * 128 + wave * 16;    // wave's query base

    const float sg    = sigma_p[0];
    const float invs2 = 1.0f / (sg * sg);

    union AB { v16bf v; unsigned int u[8]; };

    // Q A-operand (16x32 bf16): lane<16 row=cl K={0..15}; lane>=16 K={16..31}
    AB qa;
    {
        const unsigned int* Qu = (const unsigned int*)Q;
        const size_t rowbase = ((size_t)b * NN + q0 + cl) * (C2 / 2);
#pragma unroll
        for (int d = 0; d < 8; ++d) {
            int chp = h * 8 + d; // dword index in row
            qa.u[d] = Qu[rowbase + chp];
        }
    }

    v8f o0 = {0,0,0,0,0,0,0,0}, o1 = o0, o2 = o0, o3 = o0;
    float mstate[8], lstate[8];
#pragma unroll
    for (int r = 0; r < 8; ++r) { mstate[r] = -1e30f; lstate[r] = 0.0f; }

    // ---- TDM setup: descriptors + prologue load of chunk 0 into buffer 0 ----
    const u32x8 g1 = tdm_group1();
    const unsigned ldsOff0 = lds_off(&sKV[0][0]);
    const unsigned ldsOff1 = lds_off(&sKV[1][0]);
    const unsigned long long kvBase =
        (unsigned long long)(size_t)KV + (unsigned long long)b * NCHUNK * CHUNK_BYTES;

    if (wave == 0)
        tdm_load_chunk(ldsOff0, kvBase, g1);

    int buf = 0;
    for (int ci = 0; ci < NCHUNK; ++ci) {
        const int kk = ci << 5;

        if (wave == 0) {
            if (ci + 1 < NCHUNK) {   // prefetch next chunk into other buffer
                tdm_load_chunk(buf ? ldsOff0 : ldsOff1,
                               kvBase + (unsigned long long)(ci + 1) * CHUNK_BYTES, g1);
                __builtin_amdgcn_s_wait_tensorcnt(1);  // current chunk landed
            } else {
                __builtin_amdgcn_s_wait_tensorcnt(0);
            }
        }
        __syncthreads();   // publish TDM-written LDS to all waves

        const unsigned int* sKVu = (const unsigned int*)sKV[buf];

        // ---- S = Q * K^T : B-operand 32x16, lane<16 K(ch)=2v,2v+1, col=key ----
        AB kb0, kb1;
#pragma unroll
        for (int vv = 0; vv < 8; ++vv) {
            int chd = h * 8 + vv;                 // dword within key row
            kb0.u[vv] = sKVu[cl * 16 + chd];
            kb1.u[vv] = sKVu[(16 + cl) * 16 + chd];
        }
        v8f zero = {0,0,0,0,0,0,0,0};
        v8f s0 = __builtin_amdgcn_wmma_f32_16x16x32_bf16(false, qa.v, false, kb0.v,
                                                         (short)0, zero, false, false);
        v8f s1 = __builtin_amdgcn_wmma_f32_16x16x32_bf16(false, qa.v, false, kb1.v,
                                                         (short)0, zero, false, false);

        const float kq0 = Ksq[(size_t)b * NN + kk + cl];
        const float kq1 = Ksq[(size_t)b * NN + kk + 16 + cl];

        // ---- online softmax over this 32-key chunk ----
#pragma unroll
        for (int r = 0; r < 8; ++r) {
            float l0 = (s0[r] - 0.5f * kq0) * invs2;
            float l1 = (s1[r] - 0.5f * kq1) * invs2;
            float mx = fmaxf(l0, l1);
            mx = fmaxf(mx, __shfl_xor(mx, 1, 32));
            mx = fmaxf(mx, __shfl_xor(mx, 2, 32));
            mx = fmaxf(mx, __shfl_xor(mx, 4, 32));
            mx = fmaxf(mx, __shfl_xor(mx, 8, 32));
            float mnew = fmaxf(mstate[r], mx);
            float corr = __expf(mstate[r] - mnew);
            float p0 = __expf(l0 - mnew);
            float p1 = __expf(l1 - mnew);
            float rs = p0 + p1;
            rs += __shfl_xor(rs, 1, 32);
            rs += __shfl_xor(rs, 2, 32);
            rs += __shfl_xor(rs, 4, 32);
            rs += __shfl_xor(rs, 8, 32);
            lstate[r] = lstate[r] * corr + rs;
            mstate[r] = mnew;
            o0[r] *= corr; o1[r] *= corr; o2[r] *= corr; o3[r] *= corr;
            int row = r + 8 * h;                  // C-layout row of this lane
            sP[wave][row * 32 + cl]      = f2bf(p0);
            sP[wave][row * 32 + 16 + cl] = f2bf(p1);
        }

        // ---- O += P * V : P as 16x32 A-operand, Vt dwords at offset 512 ----
        AB pa;
        const unsigned int* sPu = (const unsigned int*)sP[wave];
#pragma unroll
        for (int d = 0; d < 8; ++d) {
            int kp = h * 8 + d; // key-pair dword
            pa.u[d] = sPu[cl * 16 + kp];
        }
        AB vb;
#pragma unroll
        for (int vv = 0; vv < 8; ++vv) vb.u[vv] = sKVu[512 + (0 * 16 + cl) * 16 + h * 8 + vv];
        o0 = __builtin_amdgcn_wmma_f32_16x16x32_bf16(false, pa.v, false, vb.v, (short)0, o0, false, false);
#pragma unroll
        for (int vv = 0; vv < 8; ++vv) vb.u[vv] = sKVu[512 + (1 * 16 + cl) * 16 + h * 8 + vv];
        o1 = __builtin_amdgcn_wmma_f32_16x16x32_bf16(false, pa.v, false, vb.v, (short)0, o1, false, false);
#pragma unroll
        for (int vv = 0; vv < 8; ++vv) vb.u[vv] = sKVu[512 + (2 * 16 + cl) * 16 + h * 8 + vv];
        o2 = __builtin_amdgcn_wmma_f32_16x16x32_bf16(false, pa.v, false, vb.v, (short)0, o2, false, false);
#pragma unroll
        for (int vv = 0; vv < 8; ++vv) vb.u[vv] = sKVu[512 + (3 * 16 + cl) * 16 + h * 8 + vv];
        o3 = __builtin_amdgcn_wmma_f32_16x16x32_bf16(false, pa.v, false, vb.v, (short)0, o3, false, false);

        __syncthreads();   // all waves done with buf before TDM refills it
        buf ^= 1;
    }

    // ---- normalize + residual: out flat == x flat + AO[B][N][C] flat ----
    const size_t ob = (size_t)b * ((size_t)NN * CC);
#pragma unroll
    for (int r = 0; r < 8; ++r) {
        float inv = 1.0f / lstate[r];
        int row = q0 + r + 8 * h;
        size_t base = ob + (size_t)row * CC;
        out[base + cl]      = x[base + cl]      + o0[r] * inv;
        out[base + 16 + cl] = x[base + 16 + cl] + o1[r] * inv;
        out[base + 32 + cl] = x[base + 32 + cl] + o2[r] * inv;
        out[base + 48 + cl] = x[base + 48 + cl] + o3[r] * inv;
    }
}

// ---------------------------------------------------------------------------
extern "C" void kernel_launch(void* const* d_in, const int* in_sizes, int n_in,
                              void* d_out, int out_size, void* d_ws, size_t ws_size,
                              hipStream_t stream) {
    (void)in_sizes; (void)n_in; (void)out_size; (void)ws_size;

    const float* x     = (const float*)d_in[0];
    const float* Wq    = (const float*)d_in[1];
    const float* bq    = (const float*)d_in[2];
    const float* Wk    = (const float*)d_in[3];
    const float* bk    = (const float*)d_in[4];
    const float* Wv    = (const float*)d_in[5];
    const float* bv    = (const float*)d_in[6];
    const float* sigma = (const float*)d_in[7];
    float* out = (float*)d_out;

    char* ws = (char*)d_ws;
    const size_t qBytes  = (size_t)BB * NN * C2 * 2;                 // 2 MB
    const size_t kvBytes = (size_t)BB * NCHUNK * CHUNK_BYTES;        // 6 MB
    unsigned short* Qb  = (unsigned short*)(ws);
    unsigned short* KVb = (unsigned short*)(ws + qBytes);
    float*          Ks  = (float*)(ws + qBytes + kvBytes);

    proj_kernel<<<dim3(BB * (NN / 256)), dim3(256), 0, stream>>>(
        x, Wq, bq, Wk, bk, Wv, bv, Qb, KVb, Ks);

    attn_kernel<<<dim3(BB * (NN / 128)), dim3(256), 0, stream>>>(
        x, Qb, KVb, Ks, sigma, out);
}
